// DeepGCN_75230647157385
// MI455X (gfx1250) — matrix operand order repaired and checked by reading, not compiled, not run on hardware
//
#include <hip/hip_runtime.h>
#include <hip/hip_bf16.h>

// DeepGCN (GENConv, softmax aggregation) for MI455X / gfx1250.
//  - Edge aggregation: atomic segment-max + segment-sum passes (L2-resident, 192MB L2)
//  - Node MLPs: fused WMMA f16 GEMM(128->256) -> LN -> ReLU -> WMMA GEMM(256->128)
//    with fragment-major pre-packed weights (wide 128-bit loads) and bank-conflict-free LDS.
//  - wave32 throughout.

#define NN      32768          // nodes
#define NPER    2048
#define DD      128
#define DD2     256
#define LAYERS  4
#define DEG     16
#define EE      (NN*DEG)       // edges
#define MSG_EPS 1e-7f

typedef __attribute__((ext_vector_type(16))) _Float16 v16h;
typedef __attribute__((ext_vector_type(8)))  float    v8f;

// ---------------------------------------------------------------- utilities
__device__ __forceinline__ float wave_sum32(float v) {
    #pragma unroll
    for (int o = 16; o > 0; o >>= 1) v += __shfl_xor(v, o, 32);
    return v;
}

__device__ __forceinline__ void atomic_max_f32(float* addr, float v) {
    // IEEE float ordering via int max (v>=0) / uint min (v<0); init must be -inf.
    if (v >= 0.0f) atomicMax((int*)addr, __float_as_int(v));
    else           atomicMin((unsigned int*)addr, __float_as_uint(v));
}

// A fragment: 16x32 f16, row-major source with leading dim lda (halves).
// lane 0-15: M=lane, K in {k0..k0+7, k0+16..k0+23}; lanes 16-31: same M, K+8.
__device__ __forceinline__ v16h load_a16(const _Float16* A, int lda, int k0, int lane) {
    int r = lane & 15, hs = lane >> 4;
    const _Float16* row = A + r * lda + k0 + hs * 8;
    v16h a;
    #pragma unroll
    for (int i = 0; i < 8; ++i) a[i]     = row[i];      // 16B contiguous -> ds_load_b128
    #pragma unroll
    for (int i = 0; i < 8; ++i) a[8 + i] = row[16 + i]; // 16B contiguous -> ds_load_b128
    return a;
}

// ---------------------------------------------------------------- kernels
__global__ void k_encode(const float* __restrict__ fc, const float* __restrict__ nW,
                         const float* __restrict__ nb, float* __restrict__ x) {
    int idx = blockIdx.x * blockDim.x + threadIdx.x;   // NN*32 threads
    int n = idx >> 5, q = (idx & 31) * 4;
    float c = fc[n];
    float4 w = *(const float4*)(nW + q);
    float4 b = *(const float4*)(nb + q);
    float4 r;
    r.x = c * w.x + b.x; r.y = c * w.y + b.y;
    r.z = c * w.z + b.z; r.w = c * w.w + b.w;
    *(float4*)(x + (size_t)n * DD + q) = r;
}

// Pack weights into fragment-major f16 layout:
// frag index = ((ntile*KSTEPS + kstep)*32 + lane), 16 contiguous halves per fragment:
//   element i -> K = kstep*32 + (i&7) + 8*(lane>=16) + 16*(i>>3), N = ntile*16 + (lane&15)
// W1: [128][256] (16 ntiles, 4 ksteps). W2: [256][128] (8 ntiles, 8 ksteps).
__global__ void k_pack_w(const float* __restrict__ W1, const float* __restrict__ W2,
                         _Float16* __restrict__ W1p, _Float16* __restrict__ W2p) {
    int idx = blockIdx.x * blockDim.x + threadIdx.x;   // LAYERS*32768 threads
    int l = idx >> 15;
    int r = idx & 32767;
    int i    = r & 15;
    int lane = (r >> 4) & 31;
    int c  = lane & 15, hs = lane >> 4;
    // ---- W1 pack: ks = bits[10:9], nt = bits[14:11]
    {
        int ks = (r >> 9) & 3;
        int nt = r >> 11;
        int K  = ks * 32 + (i & 7) + 8 * hs + 16 * (i >> 3);
        int Nn = nt * 16 + c;
        W1p[idx] = (_Float16)W1[(size_t)l * DD * DD2 + (size_t)K * DD2 + Nn];
    }
    // ---- W2 pack: ks = bits[11:9], nt = bits[14:12]
    {
        int ks = (r >> 9) & 7;
        int nt = r >> 12;
        int K  = ks * 32 + (i & 7) + 8 * hs + 16 * (i >> 3);
        int Nn = nt * 16 + c;
        W2p[idx] = (_Float16)W2[(size_t)l * DD2 * DD + (size_t)K * DD + Nn];
    }
}

__global__ void k_init(float* __restrict__ mx, float* __restrict__ den, float* __restrict__ num) {
    int i = (blockIdx.x * blockDim.x + threadIdx.x) * 4;
    float ninf = __int_as_float(0xff800000);
    float4 a; a.x = a.y = a.z = a.w = ninf;
    float4 z; z.x = z.y = z.z = z.w = 0.0f;
    *(float4*)(mx  + i) = a;
    *(float4*)(den + i) = z;
    *(float4*)(num + i) = z;
}

__global__ void k_edge_max(const float* __restrict__ z, const float* __restrict__ ew,
                           const int* __restrict__ src, const int* __restrict__ dst,
                           const float* __restrict__ eW, const float* __restrict__ eb,
                           const float* __restrict__ ts, int layer,
                           float* __restrict__ mx) {
    int lane = threadIdx.x & 31;
    int e = blockIdx.x * 8 + (threadIdx.x >> 5);
    float t = ts[layer];
    int s = src[e], d = dst[e];
    float w = ew[e];
    int q = lane * 4;
    float4 zz = *(const float4*)(z  + (size_t)s * DD + q);
    float4 w4 = *(const float4*)(eW + q);
    float4 b4 = *(const float4*)(eb + q);
    float m0 = fmaxf(zz.x + w * w4.x + b4.x, 0.0f) + MSG_EPS;
    float m1 = fmaxf(zz.y + w * w4.y + b4.y, 0.0f) + MSG_EPS;
    float m2 = fmaxf(zz.z + w * w4.z + b4.z, 0.0f) + MSG_EPS;
    float m3 = fmaxf(zz.w + w * w4.w + b4.w, 0.0f) + MSG_EPS;
    float* p = mx + (size_t)d * DD + q;
    atomic_max_f32(p + 0, m0 * t);
    atomic_max_f32(p + 1, m1 * t);
    atomic_max_f32(p + 2, m2 * t);
    atomic_max_f32(p + 3, m3 * t);
}

__global__ void k_edge_sum(const float* __restrict__ z, const float* __restrict__ ew,
                           const int* __restrict__ src, const int* __restrict__ dst,
                           const float* __restrict__ eW, const float* __restrict__ eb,
                           const float* __restrict__ ts, int layer,
                           const float* __restrict__ mx,
                           float* __restrict__ den, float* __restrict__ num) {
    int lane = threadIdx.x & 31;
    int e = blockIdx.x * 8 + (threadIdx.x >> 5);
    float t = ts[layer];
    int s = src[e], d = dst[e];
    float w = ew[e];
    int q = lane * 4;
    float4 zz = *(const float4*)(z  + (size_t)s * DD + q);
    float4 w4 = *(const float4*)(eW + q);
    float4 b4 = *(const float4*)(eb + q);
    float4 mv = *(const float4*)(mx + (size_t)d * DD + q);
    float m0 = fmaxf(zz.x + w * w4.x + b4.x, 0.0f) + MSG_EPS;
    float m1 = fmaxf(zz.y + w * w4.y + b4.y, 0.0f) + MSG_EPS;
    float m2 = fmaxf(zz.z + w * w4.z + b4.z, 0.0f) + MSG_EPS;
    float m3 = fmaxf(zz.w + w * w4.w + b4.w, 0.0f) + MSG_EPS;
    float e0 = __expf(m0 * t - mv.x);
    float e1 = __expf(m1 * t - mv.y);
    float e2 = __expf(m2 * t - mv.z);
    float e3 = __expf(m3 * t - mv.w);
    float* pd = den + (size_t)d * DD + q;
    float* pn = num + (size_t)d * DD + q;
    atomicAdd(pd + 0, e0);       atomicAdd(pn + 0, m0 * e0);
    atomicAdd(pd + 1, e1);       atomicAdd(pn + 1, m1 * e1);
    atomicAdd(pd + 2, e2);       atomicAdd(pn + 2, m2 * e2);
    atomicAdd(pd + 3, e3);       atomicAdd(pn + 3, m3 * e3);
}

// agg := num/(den+1e-16) + xin, in place in num
__global__ void k_fin(const float* __restrict__ den, const float* __restrict__ xin,
                      float* __restrict__ agg) {
    int i = (blockIdx.x * blockDim.x + threadIdx.x) * 4;
    float4 n4 = *(const float4*)(agg + i);
    float4 d4 = *(const float4*)(den + i);
    float4 x4 = *(const float4*)(xin + i);
    n4.x = n4.x / (d4.x + 1e-16f) + x4.x;
    n4.y = n4.y / (d4.y + 1e-16f) + x4.y;
    n4.z = n4.z / (d4.z + 1e-16f) + x4.z;
    n4.w = n4.w / (d4.w + 1e-16f) + x4.w;
    *(float4*)(agg + i) = n4;
}

// z = relu(layernorm(x, g, b)), one wave per node (D=128, 4 elems/lane)
__global__ void k_preln(const float* __restrict__ x, const float* __restrict__ g,
                        const float* __restrict__ b, float* __restrict__ z) {
    int lane = threadIdx.x & 31;
    int n = blockIdx.x * 8 + (threadIdx.x >> 5);
    int q = lane * 4;
    float4 v = *(const float4*)(x + (size_t)n * DD + q);
    float mu = wave_sum32(v.x + v.y + v.z + v.w) * (1.0f / DD);
    float dx = v.x - mu, dy = v.y - mu, dz = v.z - mu, dw = v.w - mu;
    float var = wave_sum32(dx * dx + dy * dy + dz * dz + dw * dw) * (1.0f / DD);
    float rs = rsqrtf(var + 1e-5f);
    float4 gg = *(const float4*)(g + q);
    float4 bb = *(const float4*)(b + q);
    float4 o;
    o.x = fmaxf(dx * rs * gg.x + bb.x, 0.0f);
    o.y = fmaxf(dy * rs * gg.y + bb.y, 0.0f);
    o.z = fmaxf(dz * rs * gg.z + bb.z, 0.0f);
    o.w = fmaxf(dw * rs * gg.w + bb.w, 0.0f);
    *(float4*)(z + (size_t)n * DD + q) = o;
}

// Fused MLP per 16-row tile: GEMM1 (128->256) -> +b1 -> LN -> ReLU -> GEMM2 (256->128) -> +b2 (+res)
// LDS row strides padded by 16 bytes so each of the 16 rows hits distinct bank groups.
#define HS_LD  (DD + 8)    // halves: 272B row stride
#define US_LD  (DD2 + 4)   // floats: 1040B row stride
#define UH_LD  (DD2 + 8)   // halves: 528B row stride
__global__ void __launch_bounds__(256) k_mlp(
    const float* __restrict__ h, const _Float16* __restrict__ W1p,
    const float* __restrict__ b1, const float* __restrict__ g1, const float* __restrict__ be1,
    const _Float16* __restrict__ W2p, const float* __restrict__ b2,
    const float* __restrict__ res, float* __restrict__ xout, int has_res) {
    __shared__ _Float16 Hs[16][HS_LD];
    __shared__ float    Us[16][US_LD];
    __shared__ _Float16 Uh[16][UH_LD];
    __shared__ float    redS[16][16], redQ[16][16];
    __shared__ float    muS[16], rsS[16];

    int tid = threadIdx.x;
    int wave = tid >> 5, lane = tid & 31;
    int cc = lane & 15, hs = lane >> 4;
    int row0 = blockIdx.x * 16;

    for (int i = tid; i < 16 * DD; i += 256) {
        int r = i >> 7, c = i & (DD - 1);
        Hs[r][c] = (_Float16)h[(size_t)(row0 + r) * DD + c];
    }
    __syncthreads();

    // ---- GEMM1: each wave two 16x16 tiles, cols [wave*32, wave*32+32)
    const v16h* W1f = (const v16h*)W1p;   // [(ntile*4 + ks)*32 + lane]
    int n0 = wave * 32;
    int base0 = ((2 * wave) * 4) * 32 + lane;
    int base1 = ((2 * wave + 1) * 4) * 32 + lane;
    __builtin_prefetch(W1f + base0, 0, 0);
    v8f acc0 = {}; v8f acc1 = {};
    #pragma unroll
    for (int ks = 0; ks < 4; ++ks) {
        v16h a  = load_a16(&Hs[0][0], HS_LD, ks * 32, lane);
        v16h bA = W1f[base0 + ks * 32];   // 32B contiguous -> global_load_b128 x2
        v16h bB = W1f[base1 + ks * 32];
        acc0 = __builtin_amdgcn_wmma_f32_16x16x32_f16(false, a, false, bA, (short)0, acc0, false, false);
        acc1 = __builtin_amdgcn_wmma_f32_16x16x32_f16(false, a, false, bB, (short)0, acc1, false, false);
    }
    float bias0 = b1[n0 + cc], bias1 = b1[n0 + 16 + cc];
    #pragma unroll
    for (int v = 0; v < 8; ++v) {
        int r = v + 8 * hs;
        Us[r][n0 + cc]      = acc0[v] + bias0;
        Us[r][n0 + 16 + cc] = acc1[v] + bias1;
    }
    __syncthreads();

    // ---- LayerNorm over 256 per row
    {
        int r = tid >> 4, seg = tid & 15;
        float s = 0.0f, q = 0.0f;
        #pragma unroll
        for (int i = 0; i < 16; ++i) { float vv = Us[r][seg * 16 + i]; s += vv; q += vv * vv; }
        redS[r][seg] = s; redQ[r][seg] = q;
    }
    __syncthreads();
    if (tid < 16) {
        float s = 0.0f, q = 0.0f;
        #pragma unroll
        for (int i = 0; i < 16; ++i) { s += redS[tid][i]; q += redQ[tid][i]; }
        float mu = s * (1.0f / DD2);
        float var = q * (1.0f / DD2) - mu * mu;
        muS[tid] = mu;
        rsS[tid] = rsqrtf(var + 1e-5f);
    }
    __syncthreads();
    for (int i = tid; i < 16 * DD2; i += 256) {
        int r = i >> 8, c = i & (DD2 - 1);
        float vv = (Us[r][c] - muS[r]) * rsS[r] * g1[c] + be1[c];
        Uh[r][c] = (_Float16)fmaxf(vv, 0.0f);
    }
    __syncthreads();

    // ---- GEMM2: each wave one 16x16 tile, cols [wave*16, wave*16+16)
    const v16h* W2f = (const v16h*)W2p;   // [(ntile*8 + ks)*32 + lane]
    int m0 = wave * 16;
    int base2 = (wave * 8) * 32 + lane;
    __builtin_prefetch(W2f + base2, 0, 0);
    v8f acc = {};
    #pragma unroll
    for (int ks = 0; ks < 8; ++ks) {
        v16h a = load_a16(&Uh[0][0], UH_LD, ks * 32, lane);
        v16h b = W2f[base2 + ks * 32];
        acc = __builtin_amdgcn_wmma_f32_16x16x32_f16(false, a, false, b, (short)0, acc, false, false);
    }
    float bias = b2[m0 + cc];
    #pragma unroll
    for (int v = 0; v < 8; ++v) {
        int r = v + 8 * hs;
        size_t gi = (size_t)(row0 + r) * DD + m0 + cc;
        float val = acc[v] + bias;
        if (has_res) val += res[gi];
        xout[gi] = val;
    }
}

// y[n] = dot(relu(LN(x[n], g, b)), linW) + linb
__global__ void k_head(const float* __restrict__ x, const float* __restrict__ g,
                       const float* __restrict__ b, const float* __restrict__ linW,
                       const float* __restrict__ linb, float* __restrict__ y) {
    int lane = threadIdx.x & 31;
    int n = blockIdx.x * 8 + (threadIdx.x >> 5);
    int q = lane * 4;
    float4 v = *(const float4*)(x + (size_t)n * DD + q);
    float mu = wave_sum32(v.x + v.y + v.z + v.w) * (1.0f / DD);
    float dx = v.x - mu, dy = v.y - mu, dz = v.z - mu, dw = v.w - mu;
    float var = wave_sum32(dx * dx + dy * dy + dz * dz + dw * dw) * (1.0f / DD);
    float rs = rsqrtf(var + 1e-5f);
    float4 gg = *(const float4*)(g + q);
    float4 bb = *(const float4*)(b + q);
    float4 lw = *(const float4*)(linW + q);
    float a0 = fmaxf(dx * rs * gg.x + bb.x, 0.0f);
    float a1 = fmaxf(dy * rs * gg.y + bb.y, 0.0f);
    float a2 = fmaxf(dz * rs * gg.z + bb.z, 0.0f);
    float a3 = fmaxf(dw * rs * gg.w + bb.w, 0.0f);
    float dot = wave_sum32(a0 * lw.x + a1 * lw.y + a2 * lw.z + a3 * lw.w);
    if (lane == 0) y[n] = dot + linb[0];
}

__global__ void k_sub(const float* __restrict__ y, float* __restrict__ out) {
    int n = blockIdx.x * blockDim.x + threadIdx.x;
    out[n] = y[n] - y[n & ~(NPER - 1)];
}

// ---------------------------------------------------------------- launch
extern "C" void kernel_launch(void* const* d_in, const int* in_sizes, int n_in,
                              void* d_out, int out_size, void* d_ws, size_t ws_size,
                              hipStream_t stream) {
    const float* flat_c = (const float*)d_in[0];
    const float* edge_w = (const float*)d_in[1];
    const int*   eidx   = (const int*)  d_in[2];
    const float* node_W = (const float*)d_in[3];
    const float* node_b = (const float*)d_in[4];
    const float* edge_W = (const float*)d_in[5];
    const float* edge_b = (const float*)d_in[6];
    const float* ts     = (const float*)d_in[7];
    const float* W1     = (const float*)d_in[8];
    const float* b1     = (const float*)d_in[9];
    const float* g1     = (const float*)d_in[10];
    const float* be1    = (const float*)d_in[11];
    const float* W2     = (const float*)d_in[12];
    const float* b2     = (const float*)d_in[13];
    const float* ln_g   = (const float*)d_in[14];
    const float* ln_b   = (const float*)d_in[15];
    const float* lin_W  = (const float*)d_in[16];
    const float* lin_b  = (const float*)d_in[17];
    float* out = (float*)d_out;
    (void)in_sizes; (void)n_in; (void)out_size; (void)ws_size;

    const int* src = eidx;
    const int* dst = eidx + EE;

    // workspace carve (~84.5 MB total)
    char* ws = (char*)d_ws;
    size_t off = 0;
    auto carve = [&](size_t bytes) -> char* {
        char* p = ws + off;
        off += (bytes + 255) & ~(size_t)255;
        return p;
    };
    float*    x   = (float*)   carve((size_t)NN * DD * 4);
    float*    z   = (float*)   carve((size_t)NN * DD * 4);
    float*    mx  = (float*)   carve((size_t)NN * DD * 4);
    float*    den = (float*)   carve((size_t)NN * DD * 4);
    float*    agg = (float*)   carve((size_t)NN * DD * 4);   // num, finalized in place -> MLP input
    float*    y   = (float*)   carve((size_t)NN * 4);
    _Float16* W1p = (_Float16*)carve((size_t)LAYERS * DD * DD2 * 2);
    _Float16* W2p = (_Float16*)carve((size_t)LAYERS * DD * DD2 * 2);

    k_encode<<<NN * 32 / 256, 256, 0, stream>>>(flat_c, node_W, node_b, x);
    k_pack_w<<<LAYERS * DD * DD2 / 256, 256, 0, stream>>>(W1, W2, W1p, W2p);

    for (int l = 0; l < LAYERS; ++l) {
        const float* xin = x;
        if (l > 0) {
            k_preln<<<NN / 8, 256, 0, stream>>>(x, ln_g + l * DD, ln_b + l * DD, z);
            xin = z;
        }
        k_init<<<NN * DD / 4 / 256, 256, 0, stream>>>(mx, den, agg);
        k_edge_max<<<EE / 8, 256, 0, stream>>>(xin, edge_w, src, dst, edge_W, edge_b, ts, l, mx);
        k_edge_sum<<<EE / 8, 256, 0, stream>>>(xin, edge_w, src, dst, edge_W, edge_b, ts, l,
                                               mx, den, agg);
        k_fin<<<NN * DD / 4 / 256, 256, 0, stream>>>(den, xin, agg);
        k_mlp<<<NN / 16, 256, 0, stream>>>(agg,
                                           W1p + (size_t)l * DD * DD2, b1 + l * DD2,
                                           g1 + l * DD2, be1 + l * DD2,
                                           W2p + (size_t)l * DD * DD2, b2 + l * DD,
                                           (l > 0) ? x : nullptr, x, l > 0);
    }
    k_head<<<NN / 8, 256, 0, stream>>>(x, ln_g, ln_b, lin_W, lin_b, y);
    k_sub<<<NN / 256, 256, 0, stream>>>(y, out);
}
